// FaceXHuBERT_45810121179216
// MI455X (gfx1250) — matrix operand order repaired and checked by reading, not compiled, not run on hardware
//
#include <hip/hip_runtime.h>
#include <math.h>

// Problem constants (from reference)
#define B_ 2
#define T_ 512
#define A_ 768
#define H_ 256
#define V_ 70110
#define H3_ (3*H_)        // 768
#define A2_ (2*A_)        // 1536
#define NTILES ((V_+15)/16)      // 4382 16-row v-tiles
#define VROWS (NTILES*16)        // 70112 padded fc_w rows
#define NPRED_WG ((NTILES+7)/8)  // 548 workgroups, 8 waves each
#define NRED_WG 128
#define VB 512                    // v-chunk per red workgroup iteration
#define NB ((V_+VB-1)/VB)        // 137
#define VP_ (NB*VB)              // 70144 padded v stride (multiple of 512)

typedef __attribute__((ext_vector_type(16))) __bf16 v16bf;
typedef __attribute__((ext_vector_type(8)))  float  v8f;
typedef __attribute__((ext_vector_type(4)))  unsigned int u32x4;

union BFrag { v16bf v; unsigned short u[16]; u32x4 q[2]; };

__device__ __forceinline__ unsigned short f2bf(float f) {
  unsigned int u = __float_as_uint(f);
  unsigned int r = u + 0x7fffu + ((u >> 16) & 1u);   // round-to-nearest-even
  return (unsigned short)(r >> 16);
}
__device__ __forceinline__ float bf2f(unsigned short h) {
  return __uint_as_float(((unsigned int)h) << 16);
}
__device__ __forceinline__ float wred(float x) {     // wave32 reduction
#pragma unroll
  for (int m = 16; m; m >>= 1) x += __shfl_xor(x, m, 32);
  return x;
}
__device__ __forceinline__ float sigmoidf(float x){ return 1.f/(1.f+__expf(-x)); }

__device__ __forceinline__ v8f wmma_bf(const BFrag& a, const BFrag& b, v8f c) {
  // D = A(16x32 bf16) * B(32x16 bf16) + C(16x16 f32)
  return __builtin_amdgcn_wmma_f32_16x16x32_bf16(false, a.v, false, b.v,
                                                 (short)0, c, false, false);
}

// ---------------- one-time setup kernels ----------------

// Zero GRU state, loss accumulator, and the whole srcPad matrix (rows 2..15
// and v-tail stay zero for the rest of the call; rows 0,1 (v<V_) are
// rewritten every step before being read).
__global__ void k_init(float* h0, float* h1, float* lossAcc, unsigned short* srcPad) {
  int gid = blockIdx.x*blockDim.x + threadIdx.x;
  for (int j = gid; j < B_*H_; j += gridDim.x*blockDim.x) { h0[j] = 0.f; h1[j] = 0.f; }
  for (int j = gid; j < 16*VP_; j += gridDim.x*blockDim.x) srcPad[j] = 0;
  if (gid == 0) *lossAcc = 0.f;
}

__global__ void k_f2bf(const float* __restrict__ s, unsigned short* __restrict__ d, int n) {
  for (int i = blockIdx.x*blockDim.x + threadIdx.x; i < n; i += gridDim.x*blockDim.x)
    d[i] = f2bf(s[i]);
}

// fc_w -> bf16, padded to VROWS rows (zero pad rows)
__global__ void k_cvt_fc(const float* __restrict__ s, unsigned short* __restrict__ d) {
  for (int i = blockIdx.x*blockDim.x + threadIdx.x; i < VROWS*H_; i += gridDim.x*blockDim.x) {
    int row = i / H_;
    d[i] = (row < V_) ? f2bf(s[i]) : (unsigned short)0;
  }
}

// red_w -> bf16 with padded row stride VP_ (zero pad columns)
__global__ void k_cvt_red(const float* __restrict__ s, unsigned short* __restrict__ d) {
  for (size_t i = blockIdx.x*blockDim.x + threadIdx.x; i < (size_t)A_*VP_;
       i += (size_t)gridDim.x*blockDim.x) {
    size_t a = i / VP_, v = i % VP_;
    d[i] = (v < V_) ? f2bf(s[a*V_ + v]) : (unsigned short)0;
  }
}

// src for step 0: rows 0,1 of srcPad = bf16(template)
__global__ void k_src_template(const float* __restrict__ tmpl, unsigned short* __restrict__ srcPad) {
  for (int v = blockIdx.x*blockDim.x + threadIdx.x; v < V_; v += gridDim.x*blockDim.x) {
    srcPad[v]       = f2bf(tmpl[v]);
    srcPad[VP_ + v] = f2bf(tmpl[V_ + v]);
  }
}

// ---------------- per-step kernels ----------------

// Reduce cur partials (from previous k_red), build x=[hs_t,cur] in LDS,
// compute layer-0 gate pre-activations gi0/gh0 via wave-dot products.
__global__ void __launch_bounds__(256) k_gates0(
    const float* __restrict__ hs, const float* __restrict__ red_b,
    const float* __restrict__ b_ih0, const float* __restrict__ b_hh0,
    const unsigned short* __restrict__ wih0, const unsigned short* __restrict__ whh0,
    const float* __restrict__ h0, const float* __restrict__ curPartial,
    float* __restrict__ gi0, float* __restrict__ gh0, int t) {
  __shared__ float x_lds[B_*A2_];          // [b][k], 3072 floats
  int tid = threadIdx.x;
  for (int i = tid; i < B_*A2_; i += 256) {
    int b = i / A2_, k = i % A2_;
    float val;
    if (k < A_) {
      val = hs[((size_t)b*T_ + t)*A_ + k];
    } else {
      int a = k - A_;
      float s = red_b[a];
      for (int w = 0; w < NRED_WG; ++w) s += curPartial[w*(B_*A_) + b*A_ + a];
      val = s;
    }
    x_lds[i] = val;
  }
  __syncthreads();
  int lane = tid & 31;
  int wglob = blockIdx.x*8 + (tid >> 5);   // 64 waves total
  for (int task = wglob; task < B_*H3_; task += 64) {   // gi: K=1536
    int b = task / H3_, j = task % H3_;
    const unsigned short* wr = wih0 + (size_t)j*A2_;
    float acc = 0.f;
    for (int k = lane; k < A2_; k += 32) acc += x_lds[b*A2_ + k]*bf2f(wr[k]);
    acc = wred(acc);
    if (lane == 0) gi0[task] = acc + b_ih0[j];
  }
  for (int task = wglob; task < B_*H3_; task += 64) {   // gh: K=256
    int b = task / H3_, j = task % H3_;
    const unsigned short* wr = whh0 + (size_t)j*H_;
    float acc = 0.f;
    for (int k = lane; k < H_; k += 32) acc += h0[b*H_ + k]*bf2f(wr[k]);
    acc = wred(acc);
    if (lane == 0) gh0[task] = acc + b_hh0[j];
  }
}

// Combine layer-0 cell, then full layer-1 GRU cell, single workgroup.
__global__ void __launch_bounds__(512) k_layer1(
    const float* __restrict__ gi0, const float* __restrict__ gh0,
    float* __restrict__ h0, float* __restrict__ h1,
    const unsigned short* __restrict__ wih1, const unsigned short* __restrict__ whh1,
    const float* __restrict__ b_ih1, const float* __restrict__ b_hh1) {
  __shared__ float h0n[B_*H_];
  __shared__ float g1[2*B_*H3_];           // gi1 (first half) then gh1
  int tid = threadIdx.x;
  { // layer-0 combine (512 outputs)
    int b = tid >> 8, ii = tid & 255;
    float r = sigmoidf(gi0[b*H3_+ii]        + gh0[b*H3_+ii]);
    float z = sigmoidf(gi0[b*H3_+H_+ii]     + gh0[b*H3_+H_+ii]);
    float n = tanhf   (gi0[b*H3_+2*H_+ii] + r*gh0[b*H3_+2*H_+ii]);
    float hn = (1.f - z)*n + z*h0[tid];
    h0n[tid] = hn; h0[tid] = hn;
  }
  __syncthreads();
  int lane = tid & 31, wave = tid >> 5;    // 16 waves
  for (int task = wave; task < 2*B_*H3_; task += 16) {
    int which = task / (B_*H3_);           // 0=gi1(from h0n), 1=gh1(from h1)
    int rem = task % (B_*H3_);
    int b = rem / H3_, j = rem % H3_;
    const unsigned short* wr = (which ? whh1 : wih1) + (size_t)j*H_;
    float acc = 0.f;
    if (which) { for (int k = lane; k < H_; k += 32) acc += h1[b*H_+k]*bf2f(wr[k]); }
    else       { for (int k = lane; k < H_; k += 32) acc += h0n[b*H_+k]*bf2f(wr[k]); }
    acc = wred(acc);
    if (lane == 0) g1[task] = acc + (which ? b_hh1[j] : b_ih1[j]);
  }
  __syncthreads();
  { // layer-1 combine
    int b = tid >> 8, ii = tid & 255;
    const float* gi1 = g1; const float* gh1 = g1 + B_*H3_;
    float r = sigmoidf(gi1[b*H3_+ii]        + gh1[b*H3_+ii]);
    float z = sigmoidf(gi1[b*H3_+H_+ii]     + gh1[b*H3_+H_+ii]);
    float n = tanhf   (gi1[b*H3_+2*H_+ii] + r*gh1[b*H3_+2*H_+ii]);
    h1[tid] = (1.f - z)*n + z*h1[tid];
  }
}

// pred = h1 @ fc_w.T + fc_b  (WMMA bf16), fused with MSE partial sums,
// teacher-forcing select, and bf16 src-plane stores for the red GEMM.
__global__ void __launch_bounds__(256) k_pred(
    const unsigned short* __restrict__ fcw, const float* __restrict__ fc_b,
    const float* __restrict__ h1, const float* __restrict__ vert,
    const int* __restrict__ tf_mask, float* __restrict__ out,
    unsigned short* __restrict__ srcPad, float* __restrict__ lossPartial, int t) {
  __shared__ unsigned short h1pad[16*H_];  // 16x256 bf16, cols 2..15 zero (8KB)
  __shared__ float lred[8];
  int tid = threadIdx.x;
  for (int i = tid; i < 16*H_; i += 256) {
    int col = i >> 8, k = i & 255;
    h1pad[i] = (col < 2) ? f2bf(h1[col*H_ + k]) : (unsigned short)0;
  }
  __syncthreads();
  int lane = tid & 31, wave = tid >> 5;
  int tile = blockIdx.x*8 + wave;          // one 16-v tile per wave
  int n = lane & 15, hi = lane >> 4;       // A row / B,C col = lane%16
  float lsum = 0.f;
  bool tf = tf_mask[t] != 0;
  if (tile < NTILES) {
    int v0 = tile*16;
    const unsigned short* arow = fcw + (size_t)(v0 + n)*H_;
    v8f c = {0.f,0.f,0.f,0.f,0.f,0.f,0.f,0.f};
#pragma unroll
    for (int ks = 0; ks < 8; ++ks) {       // K = 256 = 8*32
      BFrag a, bb;
      // A: 16-bit 16x32 layout (ISA 7.12.2): two aligned 16B chunks
      int kA = ks*32 + hi*8;
      a.q[0] = *(const u32x4*)(arow + kA);
      a.q[1] = *(const u32x4*)(arow + kA + 16);
      // B: column `n` (zero-padded cols), same k layout as A: two 16B chunks
      int kB = ks*32 + hi*8;
      const u32x4* bp = (const u32x4*)&h1pad[n*H_ + kB];
      bb.q[0] = bp[0];
      bb.q[1] = bp[2];
      c = wmma_bf(a, bb, c);
    }
#pragma unroll
    for (int vg = 0; vg < 8; ++vg) {       // C: lane=N(col=b), row=v0+vg(+8 hi)
      int v = v0 + vg + (hi ? 8 : 0);
      if (n < 2 && v < V_) {
        float p = c[vg] + fc_b[v];
        size_t oi = ((size_t)n*T_ + t)*(size_t)V_ + v;
        out[oi] = p;
        float gt = vert[oi];
        float d = p - gt;
        lsum += d*d;
        srcPad[(size_t)n*VP_ + v] = f2bf(tf ? gt : p);  // teacher forcing
      }
    }
  }
  lsum = wred(lsum);
  if (lane == 0) lred[wave] = lsum;
  __syncthreads();
  if (tid == 0) {
    float s = 0.f;
#pragma unroll
    for (int w = 0; w < 8; ++w) s += lred[w];
    lossPartial[blockIdx.x] = s;
  }
}

// cur partials: curPartial[wg][b][a] = sum over this wg's v-chunks of
// src[b,v]*red_w[a,v], via WMMA bf16. Also deterministically folds the
// previous k_pred's loss partials into the running loss scalar.
__global__ void __launch_bounds__(256) k_red(
    const unsigned short* __restrict__ redw, const unsigned short* __restrict__ srcPad,
    float* __restrict__ curPartial, const float* __restrict__ lossPartial,
    float* __restrict__ lossAcc, float* __restrict__ lossOut, int addLoss) {
  int tid = threadIdx.x;
  int lane = tid & 31, wave = tid >> 5;    // 8 waves
  int col = lane & 15, hi = lane >> 4;     // col = B/C col, also A row-in-tile
  v8f zero = {0.f,0.f,0.f,0.f,0.f,0.f,0.f,0.f};
  v8f c[6];
#pragma unroll
  for (int i = 0; i < 6; ++i) c[i] = zero;

  for (int vb = blockIdx.x; vb < NB; vb += NRED_WG) {
    int vbase = vb*VB;
    for (int ks = 0; ks < 16; ++ks) {      // 512 = 16*32 K per chunk
      int kv = vbase + ks*32;
      BFrag bb;
      // B fragment: src plane `col` (rows>=2 are zero), same k layout as A
      const u32x4* bp = (const u32x4*)(srcPad + (size_t)col*VP_ + kv + hi*8);
      bb.q[0] = bp[0];
      bb.q[1] = bp[2];
      __builtin_prefetch((const void*)(srcPad + (size_t)col*VP_ + kv + 512), 0, 1);
#pragma unroll
      for (int ti = 0; ti < 6; ++ti) {     // 48 a-tiles / 8 waves
        int arow = (wave + ti*8)*16 + col;
        const unsigned short* ar = redw + (size_t)arow*VP_;
        BFrag a;
        int kA = kv + hi*8;
        a.q[0] = *(const u32x4*)(ar + kA);
        a.q[1] = *(const u32x4*)(ar + kA + 16);
        c[ti] = wmma_bf(a, bb, c[ti]);
      }
    }
  }
#pragma unroll
  for (int ti = 0; ti < 6; ++ti) {
    int atile = wave + ti*8;
    if (col < 2) {
#pragma unroll
      for (int vg = 0; vg < 8; ++vg) {
        int aa = atile*16 + vg + (hi ? 8 : 0);
        curPartial[blockIdx.x*(B_*A_) + col*A_ + aa] = c[ti][vg];
      }
    }
  }
  if (blockIdx.x == 0 && tid == 0) {
    if (addLoss) {
      float s = 0.f;
      for (int i = 0; i < NPRED_WG; ++i) s += lossPartial[i];
      *lossAcc += s * (1.0f/((float)B_*(float)V_*(float)T_));
    }
    *lossOut = *lossAcc;
  }
}

// ---------------- host launcher ----------------

static inline size_t align256(size_t x){ return (x + 255) & ~(size_t)255; }

extern "C" void kernel_launch(void* const* d_in, const int* in_sizes, int n_in,
                              void* d_out, int out_size, void* d_ws, size_t ws_size,
                              hipStream_t stream) {
  const float* hs    = (const float*)d_in[0];
  const float* tmpl  = (const float*)d_in[1];
  const float* vert  = (const float*)d_in[2];
  const int*   tfm   = (const int*)d_in[3];
  const float* wih0f = (const float*)d_in[4];
  const float* whh0f = (const float*)d_in[5];
  const float* bih0  = (const float*)d_in[6];
  const float* bhh0  = (const float*)d_in[7];
  const float* wih1f = (const float*)d_in[8];
  const float* whh1f = (const float*)d_in[9];
  const float* bih1  = (const float*)d_in[10];
  const float* bhh1  = (const float*)d_in[11];
  const float* fcwf  = (const float*)d_in[12];
  const float* fcb   = (const float*)d_in[13];
  const float* redwf = (const float*)d_in[14];
  const float* redb  = (const float*)d_in[15];

  // Workspace layout (~151 MB): bf16 weight copies (L2-resident working set),
  // GRU state, padded src planes, deterministic partial buffers.
  char* ws = (char*)d_ws;
  size_t off = 0;
  auto take = [&](size_t bytes)->char* { char* p = ws + off; off = align256(off + bytes); return p; };
  unsigned short* fcw  = (unsigned short*)take((size_t)VROWS*H_*2);
  unsigned short* redw = (unsigned short*)take((size_t)A_*VP_*2);
  unsigned short* wih0 = (unsigned short*)take((size_t)H3_*A2_*2);
  unsigned short* whh0 = (unsigned short*)take((size_t)H3_*H_*2);
  unsigned short* wih1 = (unsigned short*)take((size_t)H3_*H_*2);
  unsigned short* whh1 = (unsigned short*)take((size_t)H3_*H_*2);
  unsigned short* srcPad = (unsigned short*)take((size_t)16*VP_*2);
  float* h0          = (float*)take((size_t)B_*H_*4);
  float* h1          = (float*)take((size_t)B_*H_*4);
  float* gi0         = (float*)take((size_t)B_*H3_*4);
  float* gh0         = (float*)take((size_t)B_*H3_*4);
  float* curPartial  = (float*)take((size_t)NRED_WG*B_*A_*4);
  float* lossPartial = (float*)take((size_t)NPRED_WG*4);
  float* lossAcc     = (float*)take(4);
  (void)ws_size; (void)in_sizes; (void)n_in; (void)out_size;

  float* out = (float*)d_out;
  float* lossOut = out + (size_t)B_*T_*V_;

  // One-time per call: zero state/pads, bf16-convert weights (after this the
  // whole hot weight set (~148 MB bf16) fits in the 192 MB L2 for all steps).
  k_init<<<1024,256,0,stream>>>(h0, h1, lossAcc, srcPad);
  k_cvt_fc<<<4096,256,0,stream>>>(fcwf, fcw);
  k_cvt_red<<<4096,256,0,stream>>>(redwf, redw);
  k_f2bf<<<512,256,0,stream>>>(wih0f, wih0, H3_*A2_);
  k_f2bf<<<256,256,0,stream>>>(whh0f, whh0, H3_*H_);
  k_f2bf<<<256,256,0,stream>>>(wih1f, wih1, H3_*H_);
  k_f2bf<<<256,256,0,stream>>>(whh1f, whh1, H3_*H_);

  // Step -1: cur partials from template (no loss yet)
  k_src_template<<<512,256,0,stream>>>(tmpl, srcPad);
  k_red<<<NRED_WG,256,0,stream>>>(redw, srcPad, curPartial, lossPartial, lossAcc, lossOut, 0);

  // Sequential scan over time
  for (int t = 0; t < T_; ++t) {
    k_gates0<<<8,256,0,stream>>>(hs, redb, bih0, bhh0, wih0, whh0, h0, curPartial, gi0, gh0, t);
    k_layer1<<<1,512,0,stream>>>(gi0, gh0, h0, h1, wih1, whh1, bih1, bhh1);
    k_pred<<<NPRED_WG,256,0,stream>>>(fcw, fcb, h1, vert, tfm, out, srcPad, lossPartial, t);
    k_red<<<NRED_WG,256,0,stream>>>(redw, srcPad, curPartial, lossPartial, lossAcc, lossOut, 1);
  }
}
